// Prediction_43001212567898
// MI455X (gfx1250) — matrix operand order repaired and checked by reading, not compiled, not run on hardware
//
#include <hip/hip_runtime.h>
#include <hip/hip_bf16.h>
#include <math.h>

// ---------------------------------------------------------------------------
// Problem constants (from the reference)
// ---------------------------------------------------------------------------
#define NB    256      // batch
#define NS    30       // seq len
#define LIN   14       // lstm input
#define LH    512      // lstm hidden
#define NG    2048     // 4*LH gates
#define NE    8        // experts
#define NIN   512      // expert input
#define NHID  1024     // expert hidden
#define NOUT  512      // expert output

#define LSTM_BLOCKS 32         // persistent grid: exactly (256/64) * (2048/256)

typedef __bf16 bf16;
typedef __attribute__((ext_vector_type(16))) __bf16 v16bf;
typedef __attribute__((ext_vector_type(8)))  float  v8f;
typedef unsigned int u32x4 __attribute__((ext_vector_type(4)));   // trivially-ctor 16B vector

union FragBF { v16bf v; u32x4 q[2]; };

// ---------------------------------------------------------------------------
// Elementwise helpers
// ---------------------------------------------------------------------------
__global__ void k_f32_to_bf16(const float* __restrict__ s, bf16* __restrict__ d, int n) {
    int i = blockIdx.x * blockDim.x + threadIdx.x;
    if (i < n) d[i] = (bf16)s[i];
}

__global__ void k_zero_u32(unsigned int* __restrict__ p, int n) {
    int i = blockIdx.x * blockDim.x + threadIdx.x;
    if (i < n) p[i] = 0u;
}

// ---------------------------------------------------------------------------
// 64(M) x 256(N) block tile of  C = A[M,K] * B[N,K]^T  (bf16 in, f32 out).
// 8 waves arranged 2(M) x 4(N); each wave owns 32x64 = 2x4 WMMA tiles:
// 12 b128 fragment loads per 8 v_wmma_f32_16x16x32_bf16 (1.5 loads/WMMA).
// Fragment addressing follows the CDNA5 VGPR layout tables: per lane, the A
// fragment is two contiguous 8-elem K-chunks of one row; the B fragment is the
// same two 8-elem K-chunks of one weight row. Requires K%32==0, 16B-aligned ptrs,
// uniform control flow (EXEC all-ones for WMMA).
// ---------------------------------------------------------------------------
__device__ __forceinline__ void gemm_tile_64x256(
    const bf16* __restrict__ A, const bf16* __restrict__ B, float* __restrict__ C,
    int N, int K, int rowBlk, int colBlk)
{
    const int lane = threadIdx.x & 31;
    const int wave = threadIdx.x >> 5;
    const int row0 = rowBlk + (wave >> 2) * 32;   // 2 wave-rows
    const int col0 = colBlk + (wave & 3) * 64;    // 4 wave-cols
    const int l15  = lane & 15;
    const int lhi  = lane >> 4;                   // 0 or 1

    const bf16* Ar0 = A + (size_t)(row0 + l15) * K + (lhi << 3);
    const bf16* Ar1 = Ar0 + (size_t)16 * K;
    const bf16* Bc[4];
#pragma unroll
    for (int j = 0; j < 4; ++j)
        Bc[j] = B + (size_t)(col0 + j * 16 + l15) * K + (lhi << 3);

    const v8f vzero = {0.f, 0.f, 0.f, 0.f, 0.f, 0.f, 0.f, 0.f};
    v8f acc[2][4];
#pragma unroll
    for (int i = 0; i < 2; ++i)
#pragma unroll
        for (int j = 0; j < 4; ++j) acc[i][j] = vzero;

    for (int k = 0; k < K; k += 32) {
        FragBF a0, a1, b[4];
        a0.q[0] = *(const u32x4*)(Ar0 + k);
        a0.q[1] = *(const u32x4*)(Ar0 + k + 16);
        a1.q[0] = *(const u32x4*)(Ar1 + k);
        a1.q[1] = *(const u32x4*)(Ar1 + k + 16);
#pragma unroll
        for (int j = 0; j < 4; ++j) {
            b[j].q[0] = *(const u32x4*)(Bc[j] + k);
            b[j].q[1] = *(const u32x4*)(Bc[j] + k + 16);
        }
#pragma unroll
        for (int j = 0; j < 4; ++j) {
            acc[0][j] = __builtin_amdgcn_wmma_f32_16x16x32_bf16(
                false, a0.v, false, b[j].v, (short)0, acc[0][j], false, false);
            acc[1][j] = __builtin_amdgcn_wmma_f32_16x16x32_bf16(
                false, a1.v, false, b[j].v, (short)0, acc[1][j], false, false);
        }
    }

    // C/D layout: lane<16 -> N=col+l15, VGPR v -> M=row+v; lane>=16 -> M=row+8+v.
    const int mb = row0 + lhi * 8;
    const int nb = col0 + l15;
#pragma unroll
    for (int i = 0; i < 2; ++i)
#pragma unroll
        for (int j = 0; j < 4; ++j) {
            float* cp = C + (size_t)(mb + i * 16) * N + nb + j * 16;
#pragma unroll
            for (int v = 0; v < 8; ++v) cp[(size_t)v * N] = acc[i][j][v];
        }
}

// standalone GEMM for the expert layers: grid = (N/256, M/64)
__global__ __launch_bounds__(256) void k_gemm_bf16(
    const bf16* __restrict__ A, const bf16* __restrict__ B,
    float* __restrict__ C, int N, int K)
{
    gemm_tile_64x256(A, B, C, N, K, blockIdx.y * 64, blockIdx.x * 256);
}

// ---------------------------------------------------------------------------
// Device-scope grid barrier for the persistent LSTM kernel (32 co-resident
// WGs). Release fence before arrival / acquire fence after the spin let the
// memory legalizer emit the required global_wb / global_inv for cross-WGP
// visibility on gfx1250.
// ---------------------------------------------------------------------------
__device__ __forceinline__ void gridBarrier(unsigned int* counter, unsigned int* release,
                                            unsigned int gen)
{
    __syncthreads();
    if (threadIdx.x == 0) {
        __builtin_amdgcn_fence(__ATOMIC_RELEASE, "agent");
        unsigned int arrived = __hip_atomic_fetch_add(counter, 1u, __ATOMIC_RELAXED,
                                                      __HIP_MEMORY_SCOPE_AGENT) + 1u;
        if (arrived == (unsigned int)LSTM_BLOCKS) {
            __hip_atomic_store(counter, 0u, __ATOMIC_RELAXED, __HIP_MEMORY_SCOPE_AGENT);
            __hip_atomic_fetch_add(release, 1u, __ATOMIC_RELEASE, __HIP_MEMORY_SCOPE_AGENT);
        }
        while (__hip_atomic_load(release, __ATOMIC_ACQUIRE, __HIP_MEMORY_SCOPE_AGENT) < gen) {
            __builtin_amdgcn_s_sleep(2);
        }
        __builtin_amdgcn_fence(__ATOMIC_ACQUIRE, "agent");
    }
    __syncthreads();
}

// ---------------------------------------------------------------------------
// Persistent LSTM: all 30 steps in ONE kernel (the recurrence is launch-
// latency-bound otherwise). Per step:
//   phase 1: each WG computes its fixed 64x256 tile of Yh = h@Whh^T (WMMA)
//   barrier
//   phase 2: each WG updates its fixed 4096-element (b,j) slice: fuses the
//            K=14 input projection (VALU), gate nonlinearities, and keeps the
//            cell state c ENTIRELY in registers across all 30 steps.
//   barrier
// Whh-bf16 (2 MB) stays L2-resident for the whole recurrence.
// ---------------------------------------------------------------------------
#define CELL_PER_THREAD 16     // 4096 / 256

__global__ __launch_bounds__(256) void k_lstm_persistent(
    const bf16* __restrict__ Whh_bf, const float* __restrict__ gseq,
    const float* __restrict__ Wih, const float* __restrict__ bih,
    const float* __restrict__ bhh,
    float* __restrict__ Yh, float* __restrict__ hF, bf16* __restrict__ hB,
    unsigned int* __restrict__ barCounter, unsigned int* __restrict__ barRelease)
{
    const int blk    = blockIdx.x;            // 0..31
    const int rowBlk = (blk >> 3) * 64;       // M tile (batch)
    const int colBlk = (blk & 7) * 256;       // N tile (gates)

    float creg[CELL_PER_THREAD];
#pragma unroll
    for (int t = 0; t < CELL_PER_THREAD; ++t) creg[t] = 0.f;

    unsigned int gen = 0;
    for (int s = 0; s < NS; ++s) {
        // ---- phase 1: WMMA GEMM tile of gates = h @ Whh^T ----
        gemm_tile_64x256(hB, Whh_bf, Yh, NG, LH, rowBlk, colBlk);
        gridBarrier(barCounter, barRelease, ++gen);

        // ---- phase 2: pointwise cell update on this WG's fixed slice ----
#pragma unroll
        for (int t = 0; t < CELL_PER_THREAD; ++t) {
            int idx = blk * (NB * LH / LSTM_BLOCKS) + t * 256 + threadIdx.x;
            int b = idx >> 9;        // /512
            int j = idx & 511;

            // fused input projection: x_t (14 values) vs 4 gate rows of Wih
            const float* xt = gseq + ((size_t)b * NS + s) * LIN;
            float xg0 = 0.f, xg1 = 0.f, xg2 = 0.f, xg3 = 0.f;
#pragma unroll
            for (int q = 0; q < LIN; ++q) {
                float xv = xt[q];
                xg0 += xv * Wih[(size_t)(j)          * LIN + q];
                xg1 += xv * Wih[(size_t)(LH + j)     * LIN + q];
                xg2 += xv * Wih[(size_t)(2 * LH + j) * LIN + q];
                xg3 += xv * Wih[(size_t)(3 * LH + j) * LIN + q];
            }

            const float* y = Yh + (size_t)b * NG;
            float gi = y[j]          + xg0 + bih[j]          + bhh[j];
            float gf = y[LH + j]     + xg1 + bih[LH + j]     + bhh[LH + j];
            float gg = y[2 * LH + j] + xg2 + bih[2 * LH + j] + bhh[2 * LH + j];
            float go = y[3 * LH + j] + xg3 + bih[3 * LH + j] + bhh[3 * LH + j];

            float ii = 1.f / (1.f + __expf(-gi));
            float ff = 1.f / (1.f + __expf(-gf));
            float oo = 1.f / (1.f + __expf(-go));
            float cc = ff * creg[t] + ii * tanhf(gg);
            float hh = oo * tanhf(cc);
            creg[t] = cc;
            hB[idx] = (bf16)hh;                 // next step's GEMM input
            if (s == NS - 1) hF[idx] = hh;      // final h for gating
        }
        gridBarrier(barCounter, barRelease, ++gen);
    }
}

// ---------------------------------------------------------------------------
// Gating: one wave per sample. cw = softmax(h @ Wg^T + bg), E=8.
// ---------------------------------------------------------------------------
__global__ __launch_bounds__(32) void k_gate(
    const float* __restrict__ h, const float* __restrict__ Wg,
    const float* __restrict__ bg, float* __restrict__ cw)
{
    int b = blockIdx.x;
    int lane = threadIdx.x;
    float acc[NE];
#pragma unroll
    for (int e = 0; e < NE; ++e) acc[e] = 0.f;
    for (int k = lane; k < LH; k += 32) {
        float x = h[(size_t)b * LH + k];
#pragma unroll
        for (int e = 0; e < NE; ++e) acc[e] += x * Wg[(size_t)e * LH + k];
    }
#pragma unroll
    for (int off = 16; off > 0; off >>= 1) {
#pragma unroll
        for (int e = 0; e < NE; ++e) acc[e] += __shfl_down(acc[e], off, 32);
    }
    if (lane == 0) {
        float mx = -1e30f;
#pragma unroll
        for (int e = 0; e < NE; ++e) { acc[e] += bg[e]; mx = fmaxf(mx, acc[e]); }
        float sum = 0.f;
#pragma unroll
        for (int e = 0; e < NE; ++e) { acc[e] = __expf(acc[e] - mx); sum += acc[e]; }
        float inv = 1.f / sum;
#pragma unroll
        for (int e = 0; e < NE; ++e) cw[(size_t)b * NE + e] = acc[e] * inv;
    }
}

// ---------------------------------------------------------------------------
// Expert blend:  out[b,o] = act( sum_e cw[b,e] * (Y[b, e*O + o] + bias[e,o]) )
// ---------------------------------------------------------------------------
__global__ __launch_bounds__(256) void k_blend(
    const float* __restrict__ Y, const float* __restrict__ bias,
    const float* __restrict__ cw, bf16* __restrict__ outB,
    float* __restrict__ outF, int O, int do_elu)
{
    int idx = blockIdx.x * blockDim.x + threadIdx.x;
    if (idx >= NB * O) return;
    int b = idx / O;
    int o = idx - b * O;
    const float* cwb = cw + (size_t)b * NE;
    const float* yb  = Y + (size_t)b * (NE * O) + o;
    float s = 0.f;
#pragma unroll
    for (int e = 0; e < NE; ++e)
        s += cwb[e] * (yb[(size_t)e * O] + bias[(size_t)e * O + o]);
    if (do_elu) s = (s > 0.f) ? s : expm1f(s);
    if (outB) outB[idx] = (bf16)s;
    if (outF) outF[idx] = s;
}

// ---------------------------------------------------------------------------
// Host orchestration (15 launches total)
// ---------------------------------------------------------------------------
extern "C" void kernel_launch(void* const* d_in, const int* in_sizes, int n_in,
                              void* d_out, int out_size, void* d_ws, size_t ws_size,
                              hipStream_t stream)
{
    (void)in_sizes; (void)n_in; (void)out_size; (void)ws_size;

    const float* gseq  = (const float*)d_in[0];   // [256,30,14]
    const float* xloc  = (const float*)d_in[1];   // [256,512]
    const float* Wih   = (const float*)d_in[2];   // [2048,14]
    const float* Whh   = (const float*)d_in[3];   // [2048,512]
    const float* bih   = (const float*)d_in[4];   // [2048]
    const float* bhh   = (const float*)d_in[5];   // [2048]
    const float* Wg    = (const float*)d_in[6];   // [8,512]
    const float* bg    = (const float*)d_in[7];   // [8]
    const float* W1    = (const float*)d_in[8];   // [8,1024,512]
    const float* b1    = (const float*)d_in[9];   // [8,1024]
    const float* W2    = (const float*)d_in[10];  // [8,1024,1024]
    const float* b2    = (const float*)d_in[11];  // [8,1024]
    const float* W3    = (const float*)d_in[12];  // [8,512,1024]
    const float* b3    = (const float*)d_in[13];  // [8,512]
    float* out = (float*)d_out;                   // [256,512]

    // workspace carve-up (256B aligned), ~45 MB total
    char* ws = (char*)d_ws;
    size_t off = 0;
    auto alloc = [&](size_t bytes) { char* p = ws + off; off = (off + bytes + 255) & ~(size_t)255; return p; };

    bf16*  wWhh = (bf16*)alloc((size_t)NG * LH * 2);          // 2 MB
    bf16*  wW1  = (bf16*)alloc((size_t)NE * NHID * NIN * 2);  // 8 MB
    bf16*  wW2  = (bf16*)alloc((size_t)NE * NHID * NHID * 2); // 16 MB
    bf16*  wW3  = (bf16*)alloc((size_t)NE * NOUT * NHID * 2); // 8 MB
    bf16*  xbf  = (bf16*)alloc((size_t)NB * NIN * 2);
    float* Yh   = (float*)alloc((size_t)NB * NG * 4);         // 2 MB
    float* hF   = (float*)alloc((size_t)NB * LH * 4);
    bf16*  hB   = (bf16*)alloc((size_t)NB * LH * 2);
    float* cw   = (float*)alloc((size_t)NB * NE * 4);
    float* Ybig = (float*)alloc((size_t)NB * (NE * NHID) * 4); // 8 MB (reused per layer)
    bf16*  m1   = (bf16*)alloc((size_t)NB * NHID * 2);
    bf16*  m2   = (bf16*)alloc((size_t)NB * NHID * 2);
    unsigned int* bar = (unsigned int*)alloc(256);             // [0]=counter, [1]=release

    auto cdiv = [](int a, int b) { return (a + b - 1) / b; };
    auto conv = [&](const float* s, bf16* d, int n) {
        k_f32_to_bf16<<<cdiv(n, 256), 256, 0, stream>>>(s, d, n);
    };

    // one-time per-call weight/activation bf16 conversion (~34 MB bf16, << 23.3 TB/s)
    conv(Whh,  wWhh, NG * LH);
    conv(W1,   wW1,  NE * NHID * NIN);
    conv(W2,   wW2,  NE * NHID * NHID);
    conv(W3,   wW3,  NE * NOUT * NHID);
    conv(xloc, xbf,  NB * NIN);

    // h0 = 0 (bf16); barrier state = 0 (fresh every call -> deterministic)
    k_zero_u32<<<cdiv(NB * LH / 2, 256), 256, 0, stream>>>((unsigned int*)hB, NB * LH / 2);
    k_zero_u32<<<1, 256, 0, stream>>>(bar, 2);

    // entire LSTM recurrence: ONE persistent kernel, 32 co-resident WGs
    k_lstm_persistent<<<LSTM_BLOCKS, 256, 0, stream>>>(
        wWhh, gseq, Wih, bih, bhh, Yh, hF, hB, bar, bar + 1);

    // gating softmax
    k_gate<<<NB, 32, 0, stream>>>(hF, Wg, bg, cw);

    dim3 gemmBlk(256);
    // layer 1: Y = x @ W1^T (all experts), blend+ELU -> m1 (bf16)
    dim3 gL1(NE * NHID / 256, NB / 64);   // (32, 4)
    k_gemm_bf16<<<gL1, gemmBlk, 0, stream>>>(xbf, wW1, Ybig, NE * NHID, NIN);
    k_blend<<<cdiv(NB * NHID, 256), 256, 0, stream>>>(Ybig, b1, cw, m1, nullptr, NHID, 1);

    // layer 2
    k_gemm_bf16<<<gL1, gemmBlk, 0, stream>>>(m1, wW2, Ybig, NE * NHID, NHID);
    k_blend<<<cdiv(NB * NHID, 256), 256, 0, stream>>>(Ybig, b2, cw, m2, nullptr, NHID, 1);

    // layer 3 (no ELU) -> f32 output
    dim3 gL3(NE * NOUT / 256, NB / 64);   // (16, 4)
    k_gemm_bf16<<<gL3, gemmBlk, 0, stream>>>(m2, wW3, Ybig, NE * NOUT, NHID);
    k_blend<<<cdiv(NB * NOUT, 256), 256, 0, stream>>>(Ybig, b3, cw, nullptr, out, NOUT, 0);
}
